// HierarchicalMemoryBank_850403525371
// MI455X (gfx1250) — matrix operand order repaired and checked by reading, not compiled, not run on hardware
//
#include <hip/hip_runtime.h>

// ---------------------------------------------------------------------------
// HierarchicalMemoryBank on MI455X (gfx1250): bf16 WMMA scoring + router,
// async global->LDS K-tile staging (double-buffered), register top-8,
// LDS-staged scores in bank-conflict-free layout.
// ---------------------------------------------------------------------------

typedef __attribute__((ext_vector_type(16))) __bf16        v16bf;
typedef __attribute__((ext_vector_type(8)))  float         v8f;
typedef __attribute__((ext_vector_type(8)))  unsigned int  v8u;
typedef __attribute__((ext_vector_type(4)))  unsigned int  v4u;
typedef __attribute__((ext_vector_type(4)))  int           v4i;

#define HMB_D      1024
#define HMB_DH     512
#define HMB_S      4096
#define HMB_L      3
#define HMB_TOK    16384     // B*T = 4*4096
#define HMB_TOPK   8
#define HMB_NBLK   128       // slots per score block
#define HMB_NT     (HMB_NBLK/16)
#define HMB_BSTRIDE 20       // dwords per K-tile row in LDS (16 data + 4 pad)

// gfx1250 async global->LDS path (probe-confirmed present on this toolchain;
// param1 = v4i in device/global AS, non-const — per round-2 diagnostics)
#if defined(__has_builtin)
#if __has_builtin(__builtin_amdgcn_global_load_async_to_lds_b128) && \
    __has_builtin(__builtin_amdgcn_s_wait_asynccnt)
#define HMB_ASYNC 1
#endif
#endif
#ifndef HMB_ASYNC
#define HMB_ASYNC 0
#endif

#if HMB_ASYNC
typedef __attribute__((address_space(1))) v4i hmb_as1_v4i;
typedef __attribute__((address_space(3))) v4i hmb_as3_v4i;
#endif

static __device__ __forceinline__ unsigned short hmb_f2bf(float f) {
    unsigned int u = __float_as_uint(f);
    unsigned int r = u + 0x7FFFu + ((u >> 16) & 1u);   // round-to-nearest-even
    return (unsigned short)(r >> 16);
}

// ---- fragment loaders (contiguous k order, same in A and B) ----------------------
// A 16x32 bf16: lane l: row M = l&15; VGPR v holds K pair:
//   K = 2v + 16*(l>>4)  (contiguous order)
// -> per lane two contiguous 16B chunks (compiler fuses to 2x global_load_b128)
static __device__ __forceinline__ v16bf hmb_load_a(const unsigned short* __restrict__ mat,
                                                   int row0, int k0) {
    int lane = threadIdx.x & 31;
    int l15 = lane & 15, h = lane >> 4;
    const unsigned short* rp = mat + (size_t)(row0 + l15) * HMB_D + k0;
    v8u a;
#pragma unroll
    for (int v = 0; v < 8; ++v) {
        int kp = 2 * v + 16 * h;
        a[v] = *(const unsigned int*)(rp + kp);
    }
    return __builtin_bit_cast(v16bf, a);
}

// B 32x16 bf16 frag straight from global (router path, small matrices)
static __device__ __forceinline__ v16bf hmb_load_b_global(const unsigned short* __restrict__ mat,
                                                          int row0, int k0) {
    int lane = threadIdx.x & 31;
    int l15 = lane & 15, h = lane >> 4;
    const unsigned short* rp = mat + (size_t)(row0 + l15) * HMB_D + k0 + 16 * h;
    v4u lo = *(const v4u*)rp;
    v4u hi = *(const v4u*)(rp + 8);
    v8u b = { lo[0], lo[1], lo[2], lo[3], hi[0], hi[1], hi[2], hi[3] };
    return __builtin_bit_cast(v16bf, b);
}

// ---------------------------------------------------------------------------
__global__ void hmb_cvt_bf16(const float* __restrict__ src,
                             unsigned short* __restrict__ dst, int n) {
    int i = blockIdx.x * blockDim.x + threadIdx.x;
    int stride = gridDim.x * blockDim.x;
    for (; i < n; i += stride) dst[i] = hmb_f2bf(src[i]);
}

// W1 [1024,512] (d,j) -> W1t bf16 [512,1024] (j,d)
__global__ void hmb_w1_transpose(const float* __restrict__ W1,
                                 unsigned short* __restrict__ W1t) {
    int i = blockIdx.x * blockDim.x + threadIdx.x;
    int stride = gridDim.x * blockDim.x;
    for (; i < HMB_DH * HMB_D; i += stride) {
        int j = i >> 10, d = i & (HMB_D - 1);
        W1t[i] = hmb_f2bf(W1[d * HMB_DH + j]);
    }
}

// ---------------------------------------------------------------------------
// Router: route_w[t][lvl] = softmax(relu(q@W1+b1)@W2 + b2)
__launch_bounds__(128)
__global__ void hmb_router(const unsigned short* __restrict__ qb,
                           const unsigned short* __restrict__ W1t,
                           const float* __restrict__ b1,
                           const float* __restrict__ W2,   // [512,3]
                           const float* __restrict__ b2,   // [3]
                           float* __restrict__ rw) {       // [16384,3]
    __shared__ float h_lds[4][16][17];
    __shared__ float w2_lds[HMB_DH * HMB_L];
    int tid = threadIdx.x, wave = tid >> 5, lane = tid & 31;
    int l15 = lane & 15, half = lane >> 4;
    for (int i = tid; i < HMB_DH * HMB_L; i += 128) w2_lds[i] = W2[i];
    __syncthreads();

    int trow = blockIdx.x * 64 + wave * 16;
    float lg0 = 0.f, lg1 = 0.f, lg2 = 0.f;

    for (int n0 = 0; n0 < HMB_DH; n0 += 16) {
        v8f acc = {0.f, 0.f, 0.f, 0.f, 0.f, 0.f, 0.f, 0.f};
#pragma unroll 4
        for (int k0 = 0; k0 < HMB_D; k0 += 32) {
            v16bf A = hmb_load_a(qb, trow, k0);
            v16bf B = hmb_load_b_global(W1t, n0, k0);
            acc = __builtin_amdgcn_wmma_f32_16x16x32_bf16(false, A, false, B,
                                                          (short)0, acc, false, false);
        }
        float bias = b1[n0 + l15];   // column feature j = n0 + N(=l15)
#pragma unroll
        for (int v = 0; v < 8; ++v)  // C elem (M,N) = (v+8*half, l15)
            h_lds[wave][v + 8 * half][l15] = fmaxf(acc[v] + bias, 0.f);
        __syncthreads();
        if (lane < 16) {             // lane owns token row M = lane
#pragma unroll
            for (int n = 0; n < 16; ++n) {
                float hv = h_lds[wave][lane][n];
                int j = n0 + n;
                lg0 += hv * w2_lds[j * 3 + 0];
                lg1 += hv * w2_lds[j * 3 + 1];
                lg2 += hv * w2_lds[j * 3 + 2];
            }
        }
        __syncthreads();
    }
    if (lane < 16) {
        lg0 += b2[0]; lg1 += b2[1]; lg2 += b2[2];
        float mx = fmaxf(lg0, fmaxf(lg1, lg2));
        float e0 = __expf(lg0 - mx), e1 = __expf(lg1 - mx), e2 = __expf(lg2 - mx);
        float inv = 1.f / (e0 + e1 + e2);
        int tg = trow + lane;
        rw[tg * 3 + 0] = e0 * inv;
        rw[tg * 3 + 1] = e1 * inv;
        rw[tg * 3 + 2] = e2 * inv;
    }
}

// ---------------------------------------------------------------------------
// Main attention: WG = 128 thr / 4 waves, 64 tokens, all 3 levels in-order.
// K slot-tiles staged once per WGP into LDS (async, double-buffered).
__launch_bounds__(128)
__global__ void hmb_attn(const unsigned short* __restrict__ qb,  // [16384,1024] bf16
                         const unsigned short* __restrict__ Kb,  // [3,4096,1024] bf16
                         const float* __restrict__ Vs,           // [3,4096,1024] f32
                         const float* __restrict__ sal,          // [3,4096]
                         const float* __restrict__ rw,           // [16384,3]
                         float* __restrict__ out) {              // [16384,1024]
    __shared__ float sc_lds[4][HMB_NBLK][17];          // 34.0 KB, N-major stride 17
    __shared__ unsigned int bbuf[2][HMB_NBLK * HMB_BSTRIDE];  // 20 KB K-tile dbl buffer
    __shared__ float sal_lds[HMB_NBLK];
    __shared__ float gw_lds[64][HMB_TOPK];
    __shared__ int   gi_lds[64][HMB_TOPK];

    int tid = threadIdx.x, wave = tid >> 5, lane = tid & 31;
    int l15 = lane & 15, half = lane >> 4;
    int t0 = blockIdx.x * 64;
    int trow = t0 + wave * 16;
    const float inv_sqrt_d = 0.03125f;   // 1/sqrt(1024)

    for (int lvl = 0; lvl < HMB_L; ++lvl) {
        const unsigned short* Kmat = Kb + (size_t)lvl * HMB_S * HMB_D;

        float ts[HMB_TOPK]; int ti[HMB_TOPK];
#pragma unroll
        for (int j = 0; j < HMB_TOPK; ++j) { ts[j] = -3.0e38f; ti[j] = 0; }
        float tsmin = -3.0e38f; int tsmin_pos = 0;

        for (int n0 = 0; n0 < HMB_S; n0 += HMB_NBLK) {
            sal_lds[tid] = sal[lvl * HMB_S + n0 + tid];
            // prefetch next slot-block's K rows (one line per row)
            if (n0 + HMB_NBLK < HMB_S)
                __builtin_prefetch(Kmat + (size_t)(n0 + HMB_NBLK + tid) * HMB_D, 0, 0);

            v8f acc[HMB_NT];
            v8f z = {0.f, 0.f, 0.f, 0.f, 0.f, 0.f, 0.f, 0.f};
#pragma unroll
            for (int t = 0; t < HMB_NT; ++t) acc[t] = z;

            // ---- stage K-chunk tile kc into buffer p (128 rows x 64B) ----
            // chunk c (of 512 16B pieces): row = c>>2, quarter = c&3
#define HMB_STAGE(KC, P)                                                        \
            {                                                                   \
                int kk0 = (KC) * 32;                                            \
                _Pragma("unroll")                                               \
                for (int c = tid, it = 0; it < 4; ++it, c += 128) {             \
                    int row = c >> 2, q4 = c & 3;                               \
                    const unsigned short* gp =                                  \
                        Kmat + (size_t)(n0 + row) * HMB_D + kk0 + q4 * 8;       \
                    unsigned int* lp = &bbuf[(P)][row * HMB_BSTRIDE + q4 * 4];  \
                    HMB_COPY16(gp, lp)                                          \
                }                                                               \
            }
#if HMB_ASYNC
#define HMB_COPY16(gp, lp)                                                      \
    __builtin_amdgcn_global_load_async_to_lds_b128(                             \
        (hmb_as1_v4i*)(void*)(gp), (hmb_as3_v4i*)(void*)(lp), 0, 0);
#else
#define HMB_COPY16(gp, lp) { v4u _d = *(const v4u*)(gp); *(v4u*)(lp) = _d; }
#endif

            HMB_STAGE(0, 0)
#if HMB_ASYNC
            __builtin_amdgcn_s_wait_asynccnt(0);
#endif
            __syncthreads();

            for (int kc = 0; kc < HMB_D / 32; ++kc) {
                int p = kc & 1;
                if (kc < HMB_D / 32 - 1) {
                    HMB_STAGE(kc + 1, p ^ 1)
#if HMB_ASYNC
                    __builtin_amdgcn_s_wait_asynccnt(4);  // batch kc complete
#endif
                } else {
#if HMB_ASYNC
                    __builtin_amdgcn_s_wait_asynccnt(0);
#endif
                }
                __syncthreads();

                v16bf A = hmb_load_a(qb, trow, kc * 32);
#pragma unroll
                for (int t = 0; t < HMB_NT; ++t) {
                    const unsigned int* bp =
                        &bbuf[p][(t * 16 + l15) * HMB_BSTRIDE + 8 * half];
                    v4u lo = *(const v4u*)bp;
                    v4u hi = *(const v4u*)(bp + 4);
                    v8u bu = { lo[0], lo[1], lo[2], lo[3],
                               hi[0], hi[1], hi[2], hi[3] };
                    v16bf B = __builtin_bit_cast(v16bf, bu);
                    acc[t] = __builtin_amdgcn_wmma_f32_16x16x32_bf16(
                        false, A, false, B, (short)0, acc[t], false, false);
                }
            }

            // store scores: C elem (M,N) = (v+8*half, l15); slot-local n = t*16+l15
#pragma unroll
            for (int t = 0; t < HMB_NT; ++t)
#pragma unroll
                for (int v = 0; v < 8; ++v)
                    sc_lds[wave][t * 16 + l15][v + 8 * half] = acc[t][v];
            __syncthreads();

            // scan: 2 lanes per token row (even/odd slots), running top-8
            for (int nn = 0; nn < HMB_NBLK / 2; ++nn) {
                int n = 2 * nn + half;
                float s = sc_lds[wave][n][l15] * inv_sqrt_d + sal_lds[n];
                if (s > tsmin) {
                    int idx = n0 + n;
#pragma unroll
                    for (int j = 0; j < HMB_TOPK; ++j)
                        if (j == tsmin_pos) { ts[j] = s; ti[j] = idx; }
                    tsmin = ts[0]; tsmin_pos = 0;
#pragma unroll
                    for (int j = 1; j < HMB_TOPK; ++j)
                        if (ts[j] < tsmin) { tsmin = ts[j]; tsmin_pos = j; }
                }
            }
            __syncthreads();
        }

        // merge odd-half top-8 into even-half (wave32 shfl across halves)
#pragma unroll
        for (int j = 0; j < HMB_TOPK; ++j) {
            float os = __shfl_xor(ts[j], 16, 32);
            int   oi = __shfl_xor(ti[j], 16, 32);
            if (lane < 16 && os > tsmin) {
#pragma unroll
                for (int p = 0; p < HMB_TOPK; ++p)
                    if (p == tsmin_pos) { ts[p] = os; ti[p] = oi; }
                tsmin = ts[0]; tsmin_pos = 0;
#pragma unroll
                for (int p = 1; p < HMB_TOPK; ++p)
                    if (ts[p] < tsmin) { tsmin = ts[p]; tsmin_pos = p; }
            }
        }

        // softmax over top-8, fold in route weight, publish gather table
        if (lane < 16) {
            float mx = ts[0];
#pragma unroll
            for (int j = 1; j < HMB_TOPK; ++j) mx = fmaxf(mx, ts[j]);
            float e[HMB_TOPK]; float ssum = 0.f;
#pragma unroll
            for (int j = 0; j < HMB_TOPK; ++j) { e[j] = __expf(ts[j] - mx); ssum += e[j]; }
            int tg = trow + l15;
            float w = rw[tg * 3 + lvl] / ssum;
            int tok = wave * 16 + l15;
#pragma unroll
            for (int j = 0; j < HMB_TOPK; ++j) {
                gw_lds[tok][j] = e[j] * w;
                gi_lds[tok][j] = ti[j];
            }
        }
        __syncthreads();

        // cooperative V gather: thread owns D-slice [tid*8, tid*8+8)
        const float* Vmat = Vs + (size_t)lvl * HMB_S * HMB_D;
        int d0 = tid * 8;
        for (int tok = 0; tok < 64; ++tok) {
            float r0 = 0, r1 = 0, r2 = 0, r3 = 0, r4 = 0, r5 = 0, r6 = 0, r7 = 0;
#pragma unroll
            for (int j = 0; j < HMB_TOPK; ++j) {
                float w = gw_lds[tok][j];
                const float* vp = Vmat + (size_t)gi_lds[tok][j] * HMB_D + d0;
                float4 va = *(const float4*)vp;
                float4 vb = *(const float4*)(vp + 4);
                r0 += w * va.x; r1 += w * va.y; r2 += w * va.z; r3 += w * va.w;
                r4 += w * vb.x; r5 += w * vb.y; r6 += w * vb.z; r7 += w * vb.w;
            }
            float* op = out + (size_t)(t0 + tok) * HMB_D + d0;
            if (lvl != 0) {
                float4 oa = *(const float4*)op, ob = *(const float4*)(op + 4);
                r0 += oa.x; r1 += oa.y; r2 += oa.z; r3 += oa.w;
                r4 += ob.x; r5 += ob.y; r6 += ob.z; r7 += ob.w;
            }
            float4 sa = {r0, r1, r2, r3}, sb = {r4, r5, r6, r7};
            *(float4*)op = sa;
            *(float4*)(op + 4) = sb;
        }
        __syncthreads();
    }
}

// ---------------------------------------------------------------------------
extern "C" void kernel_launch(void* const* d_in, const int* in_sizes, int n_in,
                              void* d_out, int out_size, void* d_ws, size_t ws_size,
                              hipStream_t stream) {
    const float* q   = (const float*)d_in[0];
    const float* Ks  = (const float*)d_in[1];
    const float* Vs  = (const float*)d_in[2];
    const float* sal = (const float*)d_in[3];
    const float* W1  = (const float*)d_in[4];
    const float* b1  = (const float*)d_in[5];
    const float* W2  = (const float*)d_in[6];
    const float* b2  = (const float*)d_in[7];
    float* out = (float*)d_out;

    // workspace layout (bytes): qb 32MB | Kb 24MB | W1t 1MB | rw 192KB
    char* ws = (char*)d_ws;
    unsigned short* qb  = (unsigned short*)(ws);
    unsigned short* Kb  = (unsigned short*)(ws + (size_t)33554432);
    unsigned short* W1t = (unsigned short*)(ws + (size_t)33554432 + 25165824);
    float* rw           = (float*)(ws + (size_t)33554432 + 25165824 + 1048576);

    hmb_cvt_bf16<<<2048, 256, 0, stream>>>(q, qb, HMB_TOK * HMB_D);
    hmb_cvt_bf16<<<2048, 256, 0, stream>>>(Ks, Kb, HMB_L * HMB_S * HMB_D);
    hmb_w1_transpose<<<1024, 256, 0, stream>>>(W1, W1t);
    hmb_router<<<HMB_TOK / 64, 128, 0, stream>>>(qb, W1t, b1, W2, b2, rw);
    hmb_attn<<<HMB_TOK / 64, 128, 0, stream>>>(qb, Kb, Vs, sal, rw, out);
}
